// MiniMaxText01MoE_53489522705041
// MI455X (gfx1250) — matrix operand order repaired and checked by reading, not compiled, not run on hardware
//
#include <hip/hip_runtime.h>
#include <math.h>

// ---------------- problem dims (fixed by reference) ----------------
#define T_TOK 2048
#define H_DIM 2048
#define I_DIM 2048
#define N_EXP 8
#define TOPK  2
#define R_ROWS (T_TOK * TOPK)

// ---------------- tiling ----------------
#define BM 64
#define BN 128
#define BK 32
#define LDA 40   // BK + 8 pad (16B-aligned rows, bank stagger)
#define LDB 40

typedef __attribute__((ext_vector_type(16))) __bf16 v16bf;
typedef __attribute__((ext_vector_type(8)))  __bf16 v8bf;
typedef __attribute__((ext_vector_type(4)))  __bf16 v4bf;
typedef __attribute__((ext_vector_type(8)))  float  v8f;

// Async global->LDS copy (CDNA5, tracked by ASYNCcnt). 16 bytes per lane.
#define ASYNC_B128_TO_LDS(lds_u32, gptr)                                  \
  asm volatile("global_load_async_to_lds_b128 %0, %1, off"                \
               :: "v"(lds_u32), "v"(gptr) : "memory")
#define WAIT_ASYNC0() asm volatile("s_wait_asynccnt 0" ::: "memory")

// ---- WMMA fragment loads (per CDNA5 ISA §7.12.2 16-bit layouts) ----
// A 16x32 bf16: lane<16 -> M=lane, K chunks [0..7] and [16..23];
//               lane>=16 -> same M, K chunks [8..15] and [24..31].
static __device__ __forceinline__ v16bf load_frag_a(const __bf16* base, int lane) {
  int half = lane >> 4, lr = lane & 15;
  const __bf16* row = base + lr * LDA;
  v8bf lo = *(const v8bf*)(row + half * 8);
  v8bf hi = *(const v8bf*)(row + 16 + half * 8);
  v16bf r;
#pragma unroll
  for (int i = 0; i < 8; ++i) { r[i] = lo[i]; r[i + 8] = hi[i]; }
  return r;
}
// B 32x16 bf16 (staged in LDS as Bt[N][K]): lane<16 -> N=lane, K [0..7] and [16..23];
//                                           lane>=16 -> N=lane-16, K [8..15] and [24..31].
static __device__ __forceinline__ v16bf load_frag_b(const __bf16* base, int lane) {
  int half = lane >> 4, lr = lane & 15;
  const __bf16* row = base + lr * LDB;
  v8bf lo = *(const v8bf*)(row + half * 8);
  v8bf hi = *(const v8bf*)(row + 16 + half * 8);
  v16bf r;
#pragma unroll
  for (int i = 0; i < 8; ++i) { r[i] = lo[i]; r[i + 8] = hi[i]; }
  return r;
}

// ---------------- kernel 1: fp32 -> bf16 convert of hidden_states ----------------
__global__ void cvt_bf16_kernel(const float* __restrict__ in, __bf16* __restrict__ out) {
  size_t i = ((size_t)blockIdx.x * 256 + threadIdx.x) * 4;
  float4 v = *(const float4*)(in + i);
  v4bf o;
  o[0] = (__bf16)v.x; o[1] = (__bf16)v.y; o[2] = (__bf16)v.z; o[3] = (__bf16)v.w;
  *(v4bf*)(out + i) = o;
}

// ---------------- kernel 2: zero output ----------------
__global__ void zero_out_kernel(float* __restrict__ out) {
  size_t i = ((size_t)blockIdx.x * 256 + threadIdx.x) * 4;
  float4 z = make_float4(0.f, 0.f, 0.f, 0.f);
  *(float4*)(out + i) = z;
}

// ---------------- kernel 3: router (fp32, one wave per token) ----------------
__global__ void router_kernel(const float* __restrict__ hs, const float* __restrict__ gw,
                              int* __restrict__ tki, float* __restrict__ tkw) {
  int token = blockIdx.x * 8 + (threadIdx.x >> 5);
  int lane  = threadIdx.x & 31;
  float acc[N_EXP];
#pragma unroll
  for (int e = 0; e < N_EXP; ++e) acc[e] = 0.f;
  const float* x = hs + (size_t)token * H_DIM;
  for (int h = lane * 4; h < H_DIM; h += 128) {
    float4 xv = *(const float4*)(x + h);
#pragma unroll
    for (int e = 0; e < N_EXP; ++e) {
      float4 gv = *(const float4*)(gw + (size_t)e * H_DIM + h);
      acc[e] += xv.x * gv.x + xv.y * gv.y + xv.z * gv.z + xv.w * gv.w;
    }
  }
#pragma unroll
  for (int off = 16; off >= 1; off >>= 1)
#pragma unroll
    for (int e = 0; e < N_EXP; ++e) acc[e] += __shfl_xor(acc[e], off, 32);
  if (lane == 0) {
    int i0 = 0;
#pragma unroll
    for (int e = 1; e < N_EXP; ++e) if (acc[e] > acc[i0]) i0 = e;  // first occurrence on ties
    int i1 = -1;
#pragma unroll
    for (int e = 0; e < N_EXP; ++e)
      if (e != i0 && (i1 < 0 || acc[e] > acc[i1])) i1 = e;
    // renormalized top-2 softmax weights: full denominator cancels
    float m  = acc[i0];
    float p0 = __expf(acc[i0] - m), p1 = __expf(acc[i1] - m);
    float s  = p0 + p1;
    tki[token * 2 + 0] = i0;  tki[token * 2 + 1] = i1;
    tkw[token * 2 + 0] = p0 / s;  tkw[token * 2 + 1] = p1 / s;
  }
}

// ---------------- kernel 4: deterministic routing buckets (serial) ----------------
__global__ void build_kernel(const int* __restrict__ tki, const float* __restrict__ tkw,
                             int* __restrict__ row_token, float* __restrict__ row_weight,
                             int* __restrict__ e_cnt, int* __restrict__ e_off) {
  if (threadIdx.x != 0 || blockIdx.x != 0) return;
  int cnt[N_EXP], cur[N_EXP];
#pragma unroll
  for (int e = 0; e < N_EXP; ++e) cnt[e] = 0;
  for (int r = 0; r < R_ROWS; ++r) cnt[tki[r]]++;
  int off = 0;
  for (int e = 0; e < N_EXP; ++e) { e_off[e] = off; e_cnt[e] = cnt[e]; cur[e] = off; off += cnt[e]; }
  for (int t = 0; t < T_TOK; ++t)
    for (int k = 0; k < TOPK; ++k) {
      int e = tki[t * 2 + k];
      int r = cur[e]++;
      row_token[r]  = t;
      row_weight[r] = tkw[t * 2 + k];
    }
}

// ---------------- kernel 5: grouped gate+up GEMM + SwiGLU ----------------
__global__ __launch_bounds__(256)
void moe_gateup_kernel(const __bf16* __restrict__ hsb,
                       const float* __restrict__ w1, const float* __restrict__ w3,
                       __bf16* __restrict__ act,
                       const int* __restrict__ row_token,
                       const int* __restrict__ e_cnt, const int* __restrict__ e_off) {
  int e    = blockIdx.z;
  int cnt  = e_cnt[e];
  int base = e_off[e];
  int mtile = blockIdx.y;
  if (mtile * BM >= cnt) return;
  int nbase = blockIdx.x * BN;

  __shared__ __align__(16) __bf16 As [BM * LDA];
  __shared__ __align__(16) __bf16 B1s[BN * LDB];
  __shared__ __align__(16) __bf16 B3s[BN * LDB];

  int tid = threadIdx.x, lane = tid & 31, wave = tid >> 5;
  int wm = wave >> 2, wn = wave & 3;          // 2 x 4 wave grid, 32x32 per wave
  int half = lane >> 4, lr = lane & 15;

  v8f accg[2][2], accu[2][2];
#pragma unroll
  for (int i = 0; i < 2; ++i)
#pragma unroll
    for (int j = 0; j < 2; ++j) { accg[i][j] = (v8f)0.f; accu[i][j] = (v8f)0.f; }

  // A staging: thread -> (row am, 8-elem chunk ac); gather token row, async to LDS
  int am = tid >> 2, ac = tid & 3;
  int grow_a = mtile * BM + am;
  int tok_a  = row_token[base + (grow_a < cnt ? grow_a : cnt - 1)];
  const __bf16* aptr = hsb + (size_t)tok_a * H_DIM + ac * 8;
  unsigned a_lds = (unsigned)(size_t)&As[am * LDA + ac * 8];   // LDS byte offset

  // B staging: thread owns 2 adjacent columns x 8 consecutive K rows
  int bn2 = (tid & 63) * 2;       // column pair 0..126
  int kh  = (tid >> 6) * 8;       // 0,8,16,24
  size_t woff = (size_t)e * H_DIM * I_DIM + (size_t)kh * I_DIM + nbase + bn2;
  const float* b1p = w1 + woff;
  const float* b3p = w3 + woff;

  for (int k0 = 0; k0 < H_DIM; k0 += BK) {
    ASYNC_B128_TO_LDS(a_lds, (const void*)(aptr + k0));

    const float* p1 = b1p + (size_t)k0 * I_DIM;
    const float* p3 = b3p + (size_t)k0 * I_DIM;
    float2 r1[8], r3[8];
#pragma unroll
    for (int i = 0; i < 8; ++i) {
      r1[i] = *(const float2*)(p1 + (size_t)i * I_DIM);
      r3[i] = *(const float2*)(p3 + (size_t)i * I_DIM);
    }
    v8bf c1a, c1b, c3a, c3b;
#pragma unroll
    for (int i = 0; i < 8; ++i) {
      c1a[i] = (__bf16)r1[i].x; c1b[i] = (__bf16)r1[i].y;
      c3a[i] = (__bf16)r3[i].x; c3b[i] = (__bf16)r3[i].y;
    }
    *(v8bf*)&B1s[(bn2 + 0) * LDB + kh] = c1a;
    *(v8bf*)&B1s[(bn2 + 1) * LDB + kh] = c1b;
    *(v8bf*)&B3s[(bn2 + 0) * LDB + kh] = c3a;
    *(v8bf*)&B3s[(bn2 + 1) * LDB + kh] = c3b;

    if (k0 + BK < H_DIM) {
      __builtin_prefetch(p1 + (size_t)BK * I_DIM, 0, 3);
      __builtin_prefetch(p3 + (size_t)BK * I_DIM, 0, 3);
    }
    WAIT_ASYNC0();
    __syncthreads();

    v16bf af[2];
    af[0] = load_frag_a(&As[(wm * 32 +  0) * LDA], lane);
    af[1] = load_frag_a(&As[(wm * 32 + 16) * LDA], lane);
#pragma unroll
    for (int ni = 0; ni < 2; ++ni) {
      v16bf b1f = load_frag_b(&B1s[(wn * 32 + ni * 16) * LDB], lane);
      v16bf b3f = load_frag_b(&B3s[(wn * 32 + ni * 16) * LDB], lane);
#pragma unroll
      for (int mi = 0; mi < 2; ++mi) {
        accg[mi][ni] = __builtin_amdgcn_wmma_f32_16x16x32_bf16(
            false, af[mi], false, b1f, (short)0, accg[mi][ni], false, false);
        accu[mi][ni] = __builtin_amdgcn_wmma_f32_16x16x32_bf16(
            false, af[mi], false, b3f, (short)0, accu[mi][ni], false, false);
      }
    }
    __syncthreads();
  }

  // epilogue: act = silu(g) * u  (C layout: M = msub + half*8 + r, N = lr)
#pragma unroll
  for (int mi = 0; mi < 2; ++mi) {
    int mloc = wm * 32 + mi * 16 + half * 8;
#pragma unroll
    for (int r = 0; r < 8; ++r) {
      int grow = mtile * BM + mloc + r;
      if (grow >= cnt) continue;
      size_t rowoff = (size_t)(base + grow) * I_DIM;
#pragma unroll
      for (int ni = 0; ni < 2; ++ni) {
        float g = accg[mi][ni][r], u = accu[mi][ni][r];
        float a = (g * (1.0f / (1.0f + __expf(-g)))) * u;
        act[rowoff + nbase + wn * 32 + ni * 16 + lr] = (__bf16)a;
      }
    }
  }
}

// ---------------- kernel 6: grouped down GEMM + weighted scatter ----------------
__global__ __launch_bounds__(256)
void moe_down_kernel(const __bf16* __restrict__ act, const float* __restrict__ w2,
                     float* __restrict__ out,
                     const int* __restrict__ row_token, const float* __restrict__ row_weight,
                     const int* __restrict__ e_cnt, const int* __restrict__ e_off) {
  int e    = blockIdx.z;
  int cnt  = e_cnt[e];
  int base = e_off[e];
  int mtile = blockIdx.y;
  if (mtile * BM >= cnt) return;
  int nbase = blockIdx.x * BN;

  __shared__ __align__(16) __bf16 As[BM * LDA];
  __shared__ __align__(16) __bf16 Bs[BN * LDB];

  int tid = threadIdx.x, lane = tid & 31, wave = tid >> 5;
  int wm = wave >> 2, wn = wave & 3;
  int half = lane >> 4, lr = lane & 15;

  v8f acc[2][2];
#pragma unroll
  for (int i = 0; i < 2; ++i)
#pragma unroll
    for (int j = 0; j < 2; ++j) acc[i][j] = (v8f)0.f;

  int am = tid >> 2, ac = tid & 3;
  int grow_a = mtile * BM + am;
  int arow   = base + (grow_a < cnt ? grow_a : cnt - 1);
  const __bf16* aptr = act + (size_t)arow * I_DIM + ac * 8;
  unsigned a_lds = (unsigned)(size_t)&As[am * LDA + ac * 8];

  int bn2 = (tid & 63) * 2;
  int kh  = (tid >> 6) * 8;
  const float* b2p = w2 + (size_t)e * I_DIM * H_DIM + (size_t)kh * H_DIM + nbase + bn2;

  for (int k0 = 0; k0 < I_DIM; k0 += BK) {
    ASYNC_B128_TO_LDS(a_lds, (const void*)(aptr + k0));

    const float* p2 = b2p + (size_t)k0 * H_DIM;
    float2 r2[8];
#pragma unroll
    for (int i = 0; i < 8; ++i) r2[i] = *(const float2*)(p2 + (size_t)i * H_DIM);
    v8bf c2a, c2b;
#pragma unroll
    for (int i = 0; i < 8; ++i) { c2a[i] = (__bf16)r2[i].x; c2b[i] = (__bf16)r2[i].y; }
    *(v8bf*)&Bs[(bn2 + 0) * LDB + kh] = c2a;
    *(v8bf*)&Bs[(bn2 + 1) * LDB + kh] = c2b;

    if (k0 + BK < I_DIM) __builtin_prefetch(p2 + (size_t)BK * H_DIM, 0, 3);
    WAIT_ASYNC0();
    __syncthreads();

    v16bf af[2];
    af[0] = load_frag_a(&As[(wm * 32 +  0) * LDA], lane);
    af[1] = load_frag_a(&As[(wm * 32 + 16) * LDA], lane);
#pragma unroll
    for (int ni = 0; ni < 2; ++ni) {
      v16bf bf = load_frag_b(&Bs[(wn * 32 + ni * 16) * LDB], lane);
#pragma unroll
      for (int mi = 0; mi < 2; ++mi)
        acc[mi][ni] = __builtin_amdgcn_wmma_f32_16x16x32_bf16(
            false, af[mi], false, bf, (short)0, acc[mi][ni], false, false);
    }
    __syncthreads();
  }

  // epilogue: out[token] += w * y  (exactly two adds per element -> deterministic)
#pragma unroll
  for (int mi = 0; mi < 2; ++mi) {
    int mloc = wm * 32 + mi * 16 + half * 8;
    int   toks[8]; float wts[8]; bool ok[8];
#pragma unroll
    for (int r = 0; r < 8; ++r) {
      int grow = mtile * BM + mloc + r;
      ok[r] = (grow < cnt);
      int idx = base + (ok[r] ? grow : cnt - 1);
      toks[r] = row_token[idx];
      wts[r]  = row_weight[idx];
    }
#pragma unroll
    for (int ni = 0; ni < 2; ++ni) {
      int col = nbase + wn * 32 + ni * 16 + lr;
#pragma unroll
      for (int r = 0; r < 8; ++r)
        if (ok[r])
          atomicAdd(&out[(size_t)toks[r] * H_DIM + col], wts[r] * acc[mi][ni][r]);
    }
  }
}

// ---------------- launch ----------------
extern "C" void kernel_launch(void* const* d_in, const int* in_sizes, int n_in,
                              void* d_out, int out_size, void* d_ws, size_t ws_size,
                              hipStream_t stream) {
  const float* hs = (const float*)d_in[0];
  const float* gw = (const float*)d_in[1];
  const float* w1 = (const float*)d_in[2];
  const float* w3 = (const float*)d_in[3];
  const float* w2 = (const float*)d_in[4];
  float* out = (float*)d_out;

  char* ws = (char*)d_ws;
  size_t o = 0;
  __bf16* hsb = (__bf16*)(ws + o); o += (size_t)T_TOK * H_DIM * sizeof(__bf16);
  __bf16* act = (__bf16*)(ws + o); o += (size_t)R_ROWS * I_DIM * sizeof(__bf16);
  int*    tki        = (int*)  (ws + o); o += (size_t)T_TOK * 2 * sizeof(int);
  float*  tkw        = (float*)(ws + o); o += (size_t)T_TOK * 2 * sizeof(float);
  int*    row_token  = (int*)  (ws + o); o += (size_t)R_ROWS * sizeof(int);
  float*  row_weight = (float*)(ws + o); o += (size_t)R_ROWS * sizeof(float);
  int*    e_cnt      = (int*)  (ws + o); o += 64;
  int*    e_off      = (int*)  (ws + o); o += 64;

  const int elems = T_TOK * H_DIM;  // 4M

  cvt_bf16_kernel<<<elems / 1024, 256, 0, stream>>>(hs, hsb);
  router_kernel<<<T_TOK / 8, 256, 0, stream>>>(hs, gw, tki, tkw);
  build_kernel<<<1, 32, 0, stream>>>(tki, tkw, row_token, row_weight, e_cnt, e_off);
  zero_out_kernel<<<elems / 1024, 256, 0, stream>>>(out);

  moe_gateup_kernel<<<dim3(I_DIM / BN, T_TOK / BM, N_EXP), 256, 0, stream>>>(
      hsb, w1, w3, act, row_token, e_cnt, e_off);
  moe_down_kernel<<<dim3(H_DIM / BN, T_TOK / BM, N_EXP), 256, 0, stream>>>(
      act, w2, out, row_token, row_weight, e_cnt, e_off);
}
